// DecGCN_fast_90177133346925
// MI455X (gfx1250) — compile-verified, matrix operand order checked by reading, not run on hardware
//
#include <hip/hip_runtime.h>
#include <hip/hip_bf16.h>
#include <math.h>

typedef __attribute__((ext_vector_type(16))) _Float16 v16h;
typedef __attribute__((ext_vector_type(8)))  float    v8f;
typedef __attribute__((ext_vector_type(4)))  unsigned int u32x4;
typedef __attribute__((ext_vector_type(8)))  unsigned int u32x8;

#define H     128
#define NEIGH 32
#define NDST  8192
#define NSRC  65536

// ---------------------------------------------------------------------------
// WMMA fragment swizzle (CDNA5 V_WMMA_F32_16X16X32_F16, ISA 7.12.2):
//   lane = r + 16*((k_local>>3)&1) ;  i = (k_local&7) + ((k_local&16)?8:0)
// Fragments stored pre-swizzled in LDS as [tile][lane][16 halves] so a
// fragment load is one contiguous 32-byte read per lane (ds_load_b128 x2).
// ---------------------------------------------------------------------------
__device__ __forceinline__ int swz_lane(int r, int kl) {
  return r + (((kl >> 3) & 1) << 4);
}
__device__ __forceinline__ int swz_elem(int kl) {
  return (kl & 7) + ((kl & 16) ? 8 : 0);
}
__device__ __forceinline__ v16h frag_ld(const _Float16* swzBase) {
  int lane = threadIdx.x & 31;
  return *(const v16h*)(swzBase + lane * 16);
}

// ---- K1: feat = concat(emb0[x0], emb1[x1]) @ W_in + b  -> f16 ----
__global__ __launch_bounds__(256) void feat_kernel(
    const int* __restrict__ x, const float* __restrict__ emb0,
    const float* __restrict__ emb1, const float* __restrict__ W,
    const float* __restrict__ b, _Float16* __restrict__ featH) {
  __shared__ __align__(32) _Float16 Wswz[H * H];      // [n0][k0][lane][i]
  __shared__ __align__(32) _Float16 Aswz[8 * 16 * H]; // per wave [k0][lane][i]
  __shared__ __align__(16) _Float16 Ch[8 * 16 * H];   // staged output tiles
  int t = threadIdx.x;
  for (int idx = t; idx < H * H; idx += 256) {
    int n0 = idx >> 11, r1 = idx & 2047;
    int k0 = r1 >> 9,  r2 = r1 & 511;
    int lane = r2 >> 4, i = r2 & 15;
    int kb = (lane < 16) ? 0 : 8;
    int kl = kb + ((i & 8) ? 16 : 0) + (i & 7);
    Wswz[idx] = (_Float16)W[(k0 * 32 + kl) * H + n0 * 16 + (lane & 15)];
  }
  int wave = t >> 5, lane = t & 31;
  int rowBase = blockIdx.x * 128 + wave * 16;
  _Float16* Aw = &Aswz[wave * 16 * H];
  for (int e = lane; e < 16 * H; e += 32) {
    int r = e >> 7, c = e & 127;
    int grow = rowBase + r;
    float v = (c < 32) ? emb0[(size_t)x[grow * 2 + 0] * 32 + c]
                       : emb1[(size_t)x[grow * 2 + 1] * 96 + (c - 32)];
    int k0 = c >> 5, kl = c & 31;
    Aw[(k0 * 32 + swz_lane(r, kl)) * 16 + swz_elem(kl)] = (_Float16)v;
  }
  __syncthreads();
  v16h aF[4];
#pragma unroll
  for (int k0 = 0; k0 < 4; ++k0) aF[k0] = frag_ld(&Aw[k0 * 512]);
  _Float16* Cw = &Ch[wave * 16 * H];
#pragma unroll
  for (int n0 = 0; n0 < 8; ++n0) {
    float bv = b[n0 * 16 + (lane & 15)];
    v8f acc;
#pragma unroll
    for (int e = 0; e < 8; ++e) acc[e] = bv;
#pragma unroll
    for (int k0 = 0; k0 < 4; ++k0) {
      v16h bF = frag_ld(&Wswz[n0 * 2048 + k0 * 512]);
      acc = __builtin_amdgcn_wmma_f32_16x16x32_f16(false, aF[k0], false, bF,
                                                   (short)0, acc, false, false);
    }
    int col = n0 * 16 + (lane & 15);
    int rl0 = (lane < 16) ? 0 : 8;
#pragma unroll
    for (int e = 0; e < 8; ++e) Cw[(rl0 + e) * H + col] = (_Float16)acc[e];
  }
  for (int e2 = lane; e2 < (16 * H) / 8; e2 += 32) {
    int f = e2 * 8;
    int rl = f >> 7, c = f & 127;
    uint4 val = *(const uint4*)&Cw[f];
    *(uint4*)&featH[(size_t)(rowBase + rl) * H + c] = val;
  }
}

// ---- K2: coattention pool (algebraically reduced). 1 block (4 waves) per n.
// Neighbor rows gathered via Tensor Data Mover gather-mode descriptors.
__global__ __launch_bounds__(128) void coatt_kernel(
    const _Float16* __restrict__ featH,
    const int* __restrict__ neighD, const int* __restrict__ neighQ,
    _Float16* __restrict__ rstH) {
  __shared__ __align__(16) _Float16 Ql[NEIGH * H];
  __shared__ __align__(16) _Float16 Dl[NEIGH * H];
  __shared__ __align__(32) _Float16 Dswz[NEIGH * H];
  __shared__ __align__(32) _Float16 Qswz[NEIGH * H];
  __shared__ float Lf[NEIGH * NEIGH];
  __shared__ float Mc[NEIGH], invZ[NEIGH], Mr[NEIGH], invR[NEIGH];
  __shared__ float wv[NEIGH], vv[NEIGH];
  int n = blockIdx.x;
  int t = threadIdx.x;

  // ---- TDM gather: 4 x tensor_load_to_lds, 16 indexed rows each ----
  if (t < 32) {
    unsigned long long ga = (unsigned long long)featH;
    unsigned qlds = (unsigned)(unsigned long long)(void*)Ql;
    unsigned dlds = (unsigned)(unsigned long long)(void*)Dl;
    // D# group1: mask=0, data_size=2B, dims: tensor 128 x 65536, tile 128x(16 idx)
    u32x8 g1;
    g1[0] = 0x00010000u;          // workgroup_mask=0, data_size=1 (2 bytes)
    g1[1] = 128u << 16;           // tensor_dim0 = 128 (bits 79:48, low half)
    g1[2] = 0u;                   // tensor_dim0 hi=0, tensor_dim1 lo16=0
    g1[3] = (128u << 16) | 1u;    // tensor_dim1 hi=1 (65536), tile_dim0=128
    g1[4] = 16u;                  // tile_dim1 = 16 valid indices
    g1[5] = 128u;                 // tensor_dim0_stride = 128 elements
    g1[6] = 0u;
    g1[7] = 0u;
#pragma unroll
    for (int half = 0; half < 2; ++half) {
      const int* nb = half ? neighD : neighQ;
      unsigned base = half ? dlds : qlds;
#pragma unroll
      for (int d = 0; d < 2; ++d) {
        u32x4 g0;
        g0[0] = 1u | (1u << 31);  // count=1, gather_mode=1, 16-bit indices
        g0[1] = base + (unsigned)d * 16u * (H * 2u);  // lds_addr
        g0[2] = (unsigned)(ga & 0xFFFFFFFFu);         // global_addr lo
        g0[3] = (unsigned)((ga >> 32) & 0x01FFFFFFu) | (2u << 30);  // hi | type=2
        u32x4 g2, g3;
#pragma unroll
        for (int j = 0; j < 4; ++j) {
          int r0 = d * 16 + 2 * j;
          g2[j] = ((unsigned)nb[n * NEIGH + r0] & 0xFFFFu) |
                  (((unsigned)nb[n * NEIGH + r0 + 1] & 0xFFFFu) << 16);
          g3[j] = ((unsigned)nb[n * NEIGH + r0 + 8] & 0xFFFFu) |
                  (((unsigned)nb[n * NEIGH + r0 + 9] & 0xFFFFu) << 16);
        }
        asm volatile("tensor_load_to_lds %0, %1, %2, %3"
                     :: "s"(g0), "s"(g1), "s"(g2), "s"(g3)
                     : "memory");
      }
    }
    __builtin_amdgcn_s_wait_tensorcnt(0);
  }
  __syncthreads();

  // build WMMA-swizzled copies from the gathered row-major tiles
  for (int e = t; e < 2 * NEIGH * H; e += 128) {
    int row = e >> 7;   // 0..63 : 0..31 = Q, 32..63 = D
    int c = e & 127;
    int rr = row & 31;
    int k0 = c >> 5, kl = c & 31;
    int sidx = (((rr >> 4) * 4 + k0) * 32 + swz_lane(rr & 15, kl)) * 16 + swz_elem(kl);
    if (row < NEIGH) Qswz[sidx] = Ql[rr * H + c];
    else             Dswz[sidx] = Dl[rr * H + c];
  }
  __syncthreads();
  {  // L = D Q^T : 4 waves, one 16x16 tile each, K=128
    int wave = t >> 5, lane = t & 31;
    int mi = wave >> 1, ki = wave & 1;
    v8f acc;
#pragma unroll
    for (int e = 0; e < 8; ++e) acc[e] = 0.0f;
#pragma unroll
    for (int k0 = 0; k0 < 4; ++k0) {
      v16h aF = frag_ld(&Dswz[(mi * 4 + k0) * 512]);
      v16h bF = frag_ld(&Qswz[(ki * 4 + k0) * 512]);  // B^T = rows of Q
      acc = __builtin_amdgcn_wmma_f32_16x16x32_f16(false, aF, false, bF,
                                                   (short)0, acc, false, false);
    }
    int col = ki * 16 + (lane & 15);
    int rbase = mi * 16 + ((lane < 16) ? 0 : 8);
#pragma unroll
    for (int e = 0; e < 8; ++e) Lf[(rbase + e) * NEIGH + col] = acc[e];
  }
  __syncthreads();
  if (t < NEIGH) {               // column stats (AS softmax denominators)
    int k = t;
    float mx = -1e30f;
    for (int m = 0; m < NEIGH; ++m) mx = fmaxf(mx, Lf[m * NEIGH + k]);
    float s = 0.0f;
    for (int m = 0; m < NEIGH; ++m) s += __expf(Lf[m * NEIGH + k] - mx);
    Mc[k] = mx; invZ[k] = 1.0f / s;
  } else if (t < 2 * NEIGH) {    // row stats (AC softmax denominators)
    int m = t - NEIGH;
    float mx = -1e30f;
    for (int k = 0; k < NEIGH; ++k) mx = fmaxf(mx, Lf[m * NEIGH + k]);
    float s = 0.0f;
    for (int k = 0; k < NEIGH; ++k) s += __expf(Lf[m * NEIGH + k] - mx);
    Mr[m] = mx; invR[m] = 1.0f / s;
  }
  __syncthreads();
  if (t < NEIGH) {               // w[m] = sum_k AS[k,m]
    int m = t;
    float s = 0.0f;
    for (int k = 0; k < NEIGH; ++k)
      s += __expf(Lf[m * NEIGH + k] - Mc[k]) * invZ[k];
    wv[m] = s;
  }
  __syncthreads();
  if (t < NEIGH) {               // v[k] = sum_m w[m] * AC[m,k]
    int k = t;
    float s = 0.0f;
    for (int m = 0; m < NEIGH; ++m)
      s += wv[m] * __expf(Lf[m * NEIGH + k] - Mr[m]) * invR[m];
    vv[k] = s;
  }
  __syncthreads();
  {  // channel i = t: interleaved mean of [colsumQ | w^T D | v^T Q]
    int i = t;
    float s = 0.0f;
#pragma unroll
    for (int jj = 0; jj < 3; ++jj) {
      int j = 3 * i + jj;
      float acc = 0.0f;
      if (j < H) {
        for (int r = 0; r < NEIGH; ++r) acc += (float)Ql[r * H + j];
      } else if (j < 2 * H) {
        int c = j - H;
        for (int r = 0; r < NEIGH; ++r) acc += wv[r] * (float)Dl[r * H + c];
      } else {
        int c = j - 2 * H;
        for (int r = 0; r < NEIGH; ++r) acc += vv[r] * (float)Ql[r * H + c];
      }
      s += acc;
    }
    float hself = (float)featH[(size_t)n * H + i];
    rstH[(size_t)n * H + i] = (_Float16)(hself + s * (1.0f / 96.0f));
  }
}

// ---- Generic M x 128 @ 128 x 128 WMMA GEMM with fused epilogue ----
__global__ __launch_bounds__(256) void gemm128_kernel(
    const _Float16* __restrict__ A, const float* __restrict__ W,
    const float* __restrict__ bias, float cs,
    const float* __restrict__ add0, float a0,
    const float* __restrict__ add1, float a1,
    float* __restrict__ outF, _Float16* __restrict__ outH) {
  __shared__ __align__(32) _Float16 Wswz[H * H];
  __shared__ __align__(32) _Float16 Aswz[8 * 16 * H];
  __shared__ __align__(16) float Cst[8 * 16 * H];
  int t = threadIdx.x;
  for (int idx = t; idx < H * H; idx += 256) {
    int n0 = idx >> 11, r1 = idx & 2047;
    int k0 = r1 >> 9,  r2 = r1 & 511;
    int lane = r2 >> 4, i = r2 & 15;
    int kb = (lane < 16) ? 0 : 8;
    int kl = kb + ((i & 8) ? 16 : 0) + (i & 7);
    Wswz[idx] = (_Float16)W[(k0 * 32 + kl) * H + n0 * 16 + (lane & 15)];
  }
  int wave = t >> 5, lane = t & 31;
  int rowBase = blockIdx.x * 128 + wave * 16;
  _Float16* Aw = &Aswz[wave * 16 * H];
  for (int e = lane; e < 16 * H; e += 32) {
    int r = e >> 7, c = e & 127;
    _Float16 v = A[(size_t)(rowBase + r) * H + c];
    int k0 = c >> 5, kl = c & 31;
    Aw[(k0 * 32 + swz_lane(r, kl)) * 16 + swz_elem(kl)] = v;
  }
  __syncthreads();
  v16h aF[4];
#pragma unroll
  for (int k0 = 0; k0 < 4; ++k0) aF[k0] = frag_ld(&Aw[k0 * 512]);
  float* Cw = &Cst[wave * 16 * H];
#pragma unroll
  for (int n0 = 0; n0 < 8; ++n0) {
    v8f acc;
#pragma unroll
    for (int e = 0; e < 8; ++e) acc[e] = 0.0f;
#pragma unroll
    for (int k0 = 0; k0 < 4; ++k0) {
      v16h bF = frag_ld(&Wswz[n0 * 2048 + k0 * 512]);
      acc = __builtin_amdgcn_wmma_f32_16x16x32_f16(false, aF[k0], false, bF,
                                                   (short)0, acc, false, false);
    }
    int col = n0 * 16 + (lane & 15);
    int rl0 = (lane < 16) ? 0 : 8;
#pragma unroll
    for (int e = 0; e < 8; ++e) Cw[(rl0 + e) * H + col] = acc[e];
  }
  for (int e2 = lane; e2 < (16 * H) / 2; e2 += 32) {
    int f = e2 * 2;
    int rl = f >> 7, c = f & 127;
    float2 cv = *(const float2*)&Cw[f];
    size_t gi = (size_t)(rowBase + rl) * H + c;
    float v0 = cs * cv.x, v1 = cs * cv.y;
    if (bias) { v0 += bias[c]; v1 += bias[c + 1]; }
    if (add0) { v0 += a0 * add0[gi]; v1 += a0 * add0[gi + 1]; }
    if (add1) { v0 += a1 * add1[gi]; v1 += a1 * add1[gi + 1]; }
    if (outF) { *(float2*)&outF[gi] = make_float2(v0, v1); }
    if (outH) {
      _Float16 h2[2] = {(_Float16)v0, (_Float16)v1};
      *(unsigned int*)&outH[gi] = *(unsigned int*)h2;
    }
  }
}

// ---- elementwise z1 build ----
__global__ void z1_kernel(const float* __restrict__ simF, const float* __restrict__ corF,
                          const float* __restrict__ T1, const float* __restrict__ T2,
                          _Float16* __restrict__ z1simH, _Float16* __restrict__ z1corH,
                          int total) {
  int i = blockIdx.x * blockDim.x + threadIdx.x;
  if (i < total) {
    z1simH[i] = (_Float16)(0.5f * simF[i] + 0.5f * T1[i]);
    z1corH[i] = (_Float16)(0.5f * corF[i] + 0.5f * T2[i]);
  }
}

extern "C" void kernel_launch(void* const* d_in, const int* in_sizes, int n_in,
                              void* d_out, int out_size, void* d_ws, size_t ws_size,
                              hipStream_t stream) {
  (void)in_sizes; (void)n_in; (void)out_size; (void)ws_size;
  const int*   x         = (const int*)d_in[0];
  const int*   neighS    = (const int*)d_in[1];
  const int*   neighC    = (const int*)d_in[2];
  const float* emb0_sim  = (const float*)d_in[3];
  const float* emb1_sim  = (const float*)d_in[4];
  const float* emb0_cor  = (const float*)d_in[5];
  const float* emb1_cor  = (const float*)d_in[6];
  const float* W_in_sim  = (const float*)d_in[7];
  const float* b_in_sim  = (const float*)d_in[8];
  const float* W_in_cor  = (const float*)d_in[9];
  const float* b_in_cor  = (const float*)d_in[10];
  const float* W_out_sim = (const float*)d_in[11];
  const float* b_out_sim = (const float*)d_in[12];
  const float* W_out_cor = (const float*)d_in[13];
  const float* b_out_cor = (const float*)d_in[14];
  const float* W_s2c     = (const float*)d_in[15];
  const float* W_c2s     = (const float*)d_in[16];

  char* ws = (char*)d_ws;
  size_t off = 0;
  auto carve = [&](size_t bytes) {
    char* p = ws + off;
    off += (bytes + 255) & ~(size_t)255;
    return p;
  };
  _Float16* featSimH = (_Float16*)carve((size_t)NSRC * H * 2);
  _Float16* featCorH = (_Float16*)carve((size_t)NSRC * H * 2);
  _Float16* rstSimH  = (_Float16*)carve((size_t)NDST * H * 2);
  _Float16* rstCorH  = (_Float16*)carve((size_t)NDST * H * 2);
  float*    simF     = (float*)carve((size_t)NDST * H * 4);
  float*    corF     = (float*)carve((size_t)NDST * H * 4);
  _Float16* simH     = (_Float16*)carve((size_t)NDST * H * 2);
  _Float16* corH     = (_Float16*)carve((size_t)NDST * H * 2);
  float*    T1F      = (float*)carve((size_t)NDST * H * 4);
  float*    T2F      = (float*)carve((size_t)NDST * H * 4);
  _Float16* z1simH   = (_Float16*)carve((size_t)NDST * H * 2);
  _Float16* z1corH   = (_Float16*)carve((size_t)NDST * H * 2);

  feat_kernel<<<NSRC / 128, 256, 0, stream>>>(x, emb0_sim, emb1_sim, W_in_sim, b_in_sim, featSimH);
  feat_kernel<<<NSRC / 128, 256, 0, stream>>>(x, emb0_cor, emb1_cor, W_in_cor, b_in_cor, featCorH);
  coatt_kernel<<<NDST, 128, 0, stream>>>(featSimH, neighC, neighS, rstSimH);
  coatt_kernel<<<NDST, 128, 0, stream>>>(featCorH, neighS, neighC, rstCorH);
  gemm128_kernel<<<NDST / 128, 256, 0, stream>>>(rstSimH, W_out_sim, b_out_sim, 1.0f,
                                                 nullptr, 0.0f, nullptr, 0.0f, simF, simH);
  gemm128_kernel<<<NDST / 128, 256, 0, stream>>>(rstCorH, W_out_cor, b_out_cor, 1.0f,
                                                 nullptr, 0.0f, nullptr, 0.0f, corF, corH);
  gemm128_kernel<<<NDST / 128, 256, 0, stream>>>(corH, W_c2s, nullptr, 1.0f,
                                                 nullptr, 0.0f, nullptr, 0.0f, T1F, nullptr);
  gemm128_kernel<<<NDST / 128, 256, 0, stream>>>(simH, W_s2c, nullptr, 1.0f,
                                                 nullptr, 0.0f, nullptr, 0.0f, T2F, nullptr);
  z1_kernel<<<(NDST * H + 255) / 256, 256, 0, stream>>>(simF, corF, T1F, T2F,
                                                        z1simH, z1corH, NDST * H);
  const float a2 = 0.33f;
  const float c0 = (float)(1.0 - 0.33 - 0.33);  // 0.34 as in reference
  float* z2sim = (float*)d_out;
  float* z2cor = (float*)d_out + (size_t)NDST * H;
  gemm128_kernel<<<NDST / 128, 256, 0, stream>>>(z1corH, W_c2s, nullptr, a2,
                                                 simF, c0, T1F, a2, z2sim, nullptr);
  gemm128_kernel<<<NDST / 128, 256, 0, stream>>>(z1simH, W_s2c, nullptr, a2,
                                                 corF, c0, T2F, a2, z2cor, nullptr);
}
